// Representation_Loss_63745904607779
// MI455X (gfx1250) — compile-verified, matrix operand order checked by reading, not run on hardware
//
#include <hip/hip_runtime.h>
#include <hip/hip_bf16.h>

// ---------------------------------------------------------------------------
// SimCLR NT-Xent loss, fused.  B=4096, D=256, N=8192, TEMP=0.5.
// sim = (z z^T)/TEMP via split-f16 WMMA (hi/lo), masked logsumexp done online.
//
// Grid: 64 row-groups (128 rows) x 4 col-groups (2048 cols) = 256 blocks.
// Block: 256 threads = 8 wave32s; wave w owns row tile w (16 rows, A panel in
// ~128 VGPRs); all waves share the current 16-column B panel staged in LDS by
// GLOBAL_LOAD_ASYNC_TO_LDS_B128 (double buffered, ASYNCcnt synchronized).
//
// ws: [0,4MB) zHi f16 | [4MB,8MB) zLo f16 | [8MB..) partM/S/P[4][8192] f32
// ---------------------------------------------------------------------------

typedef __attribute__((ext_vector_type(16))) _Float16 v16h;
typedef __attribute__((ext_vector_type(8)))  float    v8f;

union V16 {
    v16h  v;
    uint4 q[2];
};

#define NROW    8192
#define BHALF   4096
#define DDIM    256
#define BSTRIDE 264          // padded LDS row stride in halfs (16B aligned, bank-spread)
#define NEG_BIG (-1.0e30f)

// ---- CDNA5 async global->LDS copy (ASYNCcnt) ------------------------------
__device__ __forceinline__ void async_b128(unsigned ldsOff, const void* g) {
    asm volatile("global_load_async_to_lds_b128 %0, %1, off"
                 :: "v"(ldsOff), "v"(g)
                 : "memory");
}
__device__ __forceinline__ void wait_async0() {
    asm volatile("s_wait_asynccnt 0x0" ::: "memory");
}
// LDS aperture: flat addr low 32 bits == LDS byte offset
__device__ __forceinline__ unsigned lds_off(const void* p) {
    return (unsigned)(unsigned long long)p;
}

// ---------------------------------------------------------------------------
// Kernel 1: split f32 -> f16 hi + f16 lo (z = concat(z_i, z_j))
// ---------------------------------------------------------------------------
__global__ __launch_bounds__(256) void k_split(const float* __restrict__ zi,
                                               const float* __restrict__ zj,
                                               _Float16* __restrict__ zHi,
                                               _Float16* __restrict__ zLo) {
    int i = blockIdx.x * 256 + threadIdx.x;
    const int halfElems = BHALF * DDIM;
    float x = (i < halfElems) ? zi[i] : zj[i - halfElems];
    _Float16 h = (_Float16)x;
    zHi[i] = h;
    zLo[i] = (_Float16)(x - (float)h);
}

// ---------------------------------------------------------------------------
// Kernel 2: fused GEMM + masked online logsumexp with async-staged B panels.
// ---------------------------------------------------------------------------
__global__ __launch_bounds__(256) void k_simloss(const _Float16* __restrict__ zHi,
                                                 const _Float16* __restrict__ zLo,
                                                 float* __restrict__ pM,
                                                 float* __restrict__ pS,
                                                 float* __restrict__ pP) {
    // [buf][hi/lo][16 rows * BSTRIDE]  = 2*2*16*264*2B = 33 KB
    __shared__ _Float16 sB[2][2][16 * BSTRIDE];

    const int tid      = threadIdx.x;
    const int rowGroup = blockIdx.x >> 2;
    const int cg       = blockIdx.x & 3;

    const int wave  = tid >> 5;
    const int lane  = tid & 31;
    const int laneN = lane & 15;
    const int half  = lane >> 4;
    const int khOff = half * 8;                 // 16-bit A/B layout K-half

    const int rowTileBase = rowGroup * 128 + wave * 16;

    // ---- A panel for this wave's 16 rows -> VGPRs (hi/lo, 8 K-chunks) ----
    V16 aHi[8], aLo[8];
    {
        const int aRowG = rowTileBase + laneN;      // M = laneN
        const _Float16* aH = zHi + (size_t)aRowG * DDIM;
        const _Float16* aL = zLo + (size_t)aRowG * DDIM;
#pragma unroll
        for (int kc = 0; kc < 8; ++kc) {
            const int kb = kc * 32 + khOff;
            aHi[kc].q[0] = *(const uint4*)(aH + kb);
            aHi[kc].q[1] = *(const uint4*)(aH + kb + 16);
            aLo[kc].q[0] = *(const uint4*)(aL + kb);
            aLo[kc].q[1] = *(const uint4*)(aL + kb + 16);
        }
    }

    float m[8], s[8], p[8];
#pragma unroll
    for (int v = 0; v < 8; ++v) { m[v] = NEG_BIG; s[v] = 0.0f; p[v] = 0.0f; }

    const int ctBase = cg * 128;                // first col tile of this block

    // ---- async stage of one 16-column B panel (16 KB) into buffer b ----
    auto stage = [&](int ct, int b) {
        const int colBase = ct * 16;
        const char* gH = (const char*)(zHi + (size_t)colBase * DDIM);
        const char* gL = (const char*)(zLo + (size_t)colBase * DDIM);
#pragma unroll
        for (int i = 0; i < 2; ++i) {
            const int c   = tid * 2 + i;        // 16B chunk id, 0..511
            const int row = c >> 5;             // 0..15
            const int w   = c & 31;             // chunk within row
            const unsigned dHi = lds_off(&sB[b][0][row * BSTRIDE]) + (unsigned)(w * 16);
            const unsigned dLo = lds_off(&sB[b][1][row * BSTRIDE]) + (unsigned)(w * 16);
            async_b128(dHi, gH + (size_t)c * 16);
            async_b128(dLo, gL + (size_t)c * 16);
        }
    };

    stage(ctBase, 0);
    wait_async0();
    __syncthreads();

    for (int t = 0; t < 128; ++t) {
        const int buf = t & 1;
        if (t + 1 < 128) stage(ctBase + t + 1, buf ^ 1);   // overlap with compute

        const _Float16* bH = &sB[buf][0][laneN * BSTRIDE]; // N = laneN
        const _Float16* bL = &sB[buf][1][laneN * BSTRIDE];

        v8f acc = {};
#pragma unroll
        for (int kc = 0; kc < 8; ++kc) {
            const int kb = kc * 32 + khOff;
            V16 bHi, bLo;
            bHi.q[0] = *(const uint4*)(bH + kb);
            bHi.q[1] = *(const uint4*)(bH + kb + 16);
            bLo.q[0] = *(const uint4*)(bL + kb);
            bLo.q[1] = *(const uint4*)(bL + kb + 16);
            // split-f16: hi*hi + hi*lo + lo*hi (drop lo*lo)
            acc = __builtin_amdgcn_wmma_f32_16x16x32_f16(false, aHi[kc].v, false, bHi.v,
                                                         (short)0, acc, false, false);
            acc = __builtin_amdgcn_wmma_f32_16x16x32_f16(false, aHi[kc].v, false, bLo.v,
                                                         (short)0, acc, false, false);
            acc = __builtin_amdgcn_wmma_f32_16x16x32_f16(false, aLo[kc].v, false, bHi.v,
                                                         (short)0, acc, false, false);
        }

        // ---- mask + online softmax (per lane: one column per row v) ----
        const int colG = ctBase * 16 + t * 16 + laneN;
#pragma unroll
        for (int v = 0; v < 8; ++v) {
            const float x    = acc[v] * 2.0f;                   // /TEMP
            const int   rowG = rowTileBase + v + 8 * half;      // C layout: M = v + 8*half
            const int   d    = colG - rowG;
            const bool  isPos = (d == BHALF) || (d == -BHALF);
            const bool  valid = (d != 0) && !isPos;
            if (isPos) p[v] += x;                               // exactly one hit per row
            const float xe   = valid ? x : NEG_BIG;
            const float mNew = fmaxf(m[v], xe);
            const float w    = valid ? __expf(x - mNew) : 0.0f;
            s[v] = s[v] * __expf(m[v] - mNew) + w;
            m[v] = mNew;
        }

        wait_async0();       // next panel fully in LDS
        __syncthreads();     // all waves done with both buffers' hand-off
    }

    // ---- cross-lane merge within each 16-lane half (C-layout columns) ----
#pragma unroll
    for (int v = 0; v < 8; ++v) {
#pragma unroll
        for (int off = 1; off < 16; off <<= 1) {
            const float mo = __shfl_xor(m[v], off, 32);
            const float so = __shfl_xor(s[v], off, 32);
            const float po = __shfl_xor(p[v], off, 32);
            const float mN = fmaxf(m[v], mo);
            s[v] = s[v] * __expf(m[v] - mN) + so * __expf(mo - mN);
            m[v] = mN;
            p[v] += po;
        }
    }

    // ---- per-row column-group partials -> workspace ----
    if (laneN == 0) {
#pragma unroll
        for (int v = 0; v < 8; ++v) {
            const int r = rowTileBase + v + 8 * half;
            pM[cg * NROW + r] = m[v];
            pS[cg * NROW + r] = s[v];
            pP[cg * NROW + r] = p[v];
        }
    }
}

// ---------------------------------------------------------------------------
// Kernel 3: merge 4 column-group partials per row, finish lse, deterministic
// reduction: loss = sum(lse - pos) / N^2
// ---------------------------------------------------------------------------
__global__ __launch_bounds__(256) void k_final(const float* __restrict__ pM,
                                               const float* __restrict__ pS,
                                               const float* __restrict__ pP,
                                               float* __restrict__ out) {
    __shared__ float buf[256];
    float acc = 0.0f;
    for (int r = threadIdx.x; r < NROW; r += 256) {
        float M = pM[r], S = pS[r], P = pP[r];
#pragma unroll
        for (int c = 1; c < 4; ++c) {
            const float mo = pM[c * NROW + r], so = pS[c * NROW + r];
            const float mN = fmaxf(M, mo);
            S = S * __expf(M - mN) + so * __expf(mo - mN);
            M = mN;
            P += pP[c * NROW + r];
        }
        const float mN  = fmaxf(M, P);                 // logits = [pos | negs]
        const float tot = S * __expf(M - mN) + __expf(P - mN);
        acc += (mN + __logf(tot)) - P;
    }
    buf[threadIdx.x] = acc;
    __syncthreads();
    for (int st = 128; st > 0; st >>= 1) {
        if (threadIdx.x < st) buf[threadIdx.x] += buf[threadIdx.x + st];
        __syncthreads();
    }
    if (threadIdx.x == 0)
        out[0] = buf[0] * (1.0f / ((float)NROW * (float)NROW));
}

// ---------------------------------------------------------------------------
extern "C" void kernel_launch(void* const* d_in, const int* in_sizes, int n_in,
                              void* d_out, int out_size, void* d_ws, size_t ws_size,
                              hipStream_t stream) {
    (void)in_sizes; (void)n_in; (void)out_size; (void)ws_size;
    const float* zi = (const float*)d_in[0];
    const float* zj = (const float*)d_in[1];

    _Float16* zHi = (_Float16*)d_ws;
    _Float16* zLo = (_Float16*)((char*)d_ws + (size_t)4 * 1024 * 1024);
    float*    pM  = (float*)((char*)d_ws + (size_t)8 * 1024 * 1024);
    float*    pS  = pM + 4 * NROW;
    float*    pP  = pS + 4 * NROW;
    float*    out = (float*)d_out;

    // 1) split z (concat) into f16 hi/lo
    k_split<<<(NROW * DDIM) / 256, 256, 0, stream>>>(zi, zj, zHi, zLo);
    // 2) fused GEMM + masked online logsumexp (256 blocks x 256 threads)
    k_simloss<<<256, 256, 0, stream>>>(zHi, zLo, pM, pS, pP);
    // 3) merge partials + deterministic scalar reduction
    k_final<<<1, 256, 0, stream>>>(pM, pS, pP, out);
}
